// ASTGCN_block__3513283248985
// MI455X (gfx1250) — matrix operand order repaired and checked by reading, not compiled, not run on hardware
//
#include <hip/hip_runtime.h>
#include <hip/hip_bf16.h>
#include <stdint.h>

// Problem constants (B,N,C,T = 32,512,64,24 ; K=3 ; FC=FT=64)
#define BB  32
#define NN  512
#define CC  64
#define TT  24
#define KC  3
#define FCC 64
#define FTT 64
#define KM  (KC * NN)    // 1536  (fused k*m contraction for cheb GEMM)
#define FT2 (FCC * TT)   // 1536  (fused f*t output columns for cheb GEMM)
#define LN_EPS 1e-5f

typedef __attribute__((ext_vector_type(16))) __bf16 v16bf;
typedef __attribute__((ext_vector_type(8)))  __bf16 v8bf;
typedef __attribute__((ext_vector_type(8)))  float  v8f;

static __device__ __forceinline__ unsigned short f2bf(float f) {
  unsigned u = __float_as_uint(f);
  u += 0x7FFFu + ((u >> 16) & 1u);   // round-to-nearest-even
  return (unsigned short)(u >> 16);
}
static __device__ __forceinline__ float sigm(float v) {
  return 1.0f / (1.0f + expf(-v));
}

// Build a 16-element bf16 fragment from two contiguous 8-element (16B) chunks.
// Emits two global_load_b128 instead of 16 strided u16 loads.
static __device__ __forceinline__ v16bf load_frag16(const unsigned short* p0,
                                                    const unsigned short* p1) {
  union { uint4 u; v8bf h; } lo, hi;
  lo.u = *(const uint4*)p0;
  hi.u = *(const uint4*)p1;
  v16bf r;
#pragma unroll
  for (int j = 0; j < 8; ++j) { r[j] = lo.h[j]; r[j + 8] = hi.h[j]; }
  return r;
}

// ---------------- temporal attention ----------------
// lhs_t[b,t,c] = sum_n x[b,n,c,t] * U1[n]
__global__ void k_lhs_t(const float* __restrict__ x, const float* __restrict__ U1,
                        float* __restrict__ lhs) {
  int i = blockIdx.x * blockDim.x + threadIdx.x;
  if (i >= BB * TT * CC) return;
  int c = i % CC, t = (i / CC) % TT, b = i / (CC * TT);
  const float* xp = x + (size_t)b * NN * CC * TT + (size_t)c * TT + t;
  float s = 0.f;
  for (int n = 0; n < NN; ++n) s += xp[(size_t)n * CC * TT] * U1[n];
  lhs[i] = s;
}

// lhs2[b,t,n] = sum_c lhs_t[b,t,c] * U2[c,n]
__global__ void k_lhs_t2(const float* __restrict__ lhs, const float* __restrict__ U2,
                         float* __restrict__ lhs2) {
  int i = blockIdx.x * blockDim.x + threadIdx.x;
  if (i >= BB * TT * NN) return;
  int n = i % NN, bt = i / NN;
  const float* lp = lhs + (size_t)bt * CC;
  float s = 0.f;
  for (int c = 0; c < CC; ++c) s += lp[c] * U2[(size_t)c * NN + n];
  lhs2[i] = s;
}

// rhs_t[b,n,t] = sum_c U3[c] * x[b,n,c,t]
__global__ void k_rhs_t(const float* __restrict__ x, const float* __restrict__ U3,
                        float* __restrict__ rhs) {
  int i = blockIdx.x * blockDim.x + threadIdx.x;
  if (i >= BB * NN * TT) return;
  int t = i % TT, bn = i / TT;
  const float* xp = x + (size_t)bn * CC * TT + t;
  float s = 0.f;
  for (int c = 0; c < CC; ++c) s += U3[c] * xp[(size_t)c * TT];
  rhs[i] = s;
}

// prodsig[b,t,u] = sigmoid( sum_n lhs2[b,t,n]*rhs_t[b,n,u] + be[t,u] )
__global__ void k_prodsig_t(const float* __restrict__ lhs2, const float* __restrict__ rhs,
                            const float* __restrict__ be, float* __restrict__ ps) {
  int i = blockIdx.x * blockDim.x + threadIdx.x;
  if (i >= BB * TT * TT) return;
  int u = i % TT, t = (i / TT) % TT, b = i / (TT * TT);
  const float* lp = lhs2 + ((size_t)b * TT + t) * NN;
  const float* rp = rhs + (size_t)b * NN * TT + u;
  float s = 0.f;
  for (int n = 0; n < NN; ++n) s += lp[n] * rp[(size_t)n * TT];
  ps[i] = sigm(s + be[t * TT + u]);
}

// E[b,t,u] = softmax_t( sum_s Ve[t,s]*ps[b,s,u] )
__global__ void k_E(const float* __restrict__ Ve, const float* __restrict__ ps,
                    float* __restrict__ E) {
  int i = blockIdx.x * blockDim.x + threadIdx.x;
  if (i >= BB * TT) return;
  int u = i % TT, b = i / TT;
  float e[TT];
  float mx = -1e30f;
  for (int t = 0; t < TT; ++t) {
    float s = 0.f;
    for (int ss = 0; ss < TT; ++ss)
      s += Ve[t * TT + ss] * ps[((size_t)b * TT + ss) * TT + u];
    e[t] = s; mx = fmaxf(mx, s);
  }
  float sm = 0.f;
  for (int t = 0; t < TT; ++t) { e[t] = expf(e[t] - mx); sm += e[t]; }
  float inv = 1.f / sm;
  for (int t = 0; t < TT; ++t) E[((size_t)b * TT + t) * TT + u] = e[t] * inv;
}

// x_TAt[b,n,c,t] = sum_s x[b,n,c,s] * E[b,s,t]
__global__ void k_xTAt(const float* __restrict__ x, const float* __restrict__ E,
                       float* __restrict__ y) {
  int i = blockIdx.x * blockDim.x + threadIdx.x;
  if (i >= BB * NN * CC * TT) return;
  int t = i % TT;
  size_t bnc = (size_t)(i / TT);
  int b = (int)(bnc / (NN * CC));
  const float* xr = x + bnc * TT;
  const float* Ep = E + (size_t)b * TT * TT + t;
  float s = 0.f;
  for (int ss = 0; ss < TT; ++ss) s += xr[ss] * Ep[(size_t)ss * TT];
  y[i] = s;
}

// ---------------- spatial attention ----------------
__global__ void k_lhs_s(const float* __restrict__ xa, const float* __restrict__ W1,
                        float* __restrict__ lhs) {
  int i = blockIdx.x * blockDim.x + threadIdx.x;
  if (i >= BB * NN * CC) return;
  const float* xp = xa + (size_t)i * TT;
  float s = 0.f;
  for (int t = 0; t < TT; ++t) s += xp[t] * W1[t];
  lhs[i] = s;
}

__global__ void k_lhs_s2(const float* __restrict__ lhs, const float* __restrict__ W2,
                         float* __restrict__ lhs2) {
  int i = blockIdx.x * blockDim.x + threadIdx.x;
  if (i >= BB * NN * TT) return;
  int t = i % TT, bn = i / TT;
  const float* lp = lhs + (size_t)bn * CC;
  float s = 0.f;
  for (int c = 0; c < CC; ++c) s += lp[c] * W2[c * TT + t];
  lhs2[i] = s;
}

__global__ void k_rhs_s(const float* __restrict__ xa, const float* __restrict__ W3,
                        float* __restrict__ rhs) {
  int i = blockIdx.x * blockDim.x + threadIdx.x;
  if (i >= BB * NN * TT) return;
  int t = i % TT, bn = i / TT;
  const float* xp = xa + (size_t)bn * CC * TT + t;
  float s = 0.f;
  for (int c = 0; c < CC; ++c) s += W3[c] * xp[(size_t)c * TT];
  rhs[i] = s;
}

// sigT[b,q,m] = bf16( sigmoid( lhs2[b,m,:]·rhs[b,q,:] + bs[m,q] ) )
// Stored TRANSPOSED (contraction index m contiguous) so the GEMM B-fragment
// loads are two contiguous b128s per tile.
__global__ void k_prodsig_s(const float* __restrict__ lhs2, const float* __restrict__ rhs,
                            const float* __restrict__ bs, unsigned short* __restrict__ out) {
  int i = blockIdx.x * blockDim.x + threadIdx.x;
  if (i >= BB * NN * NN) return;
  int m = i % NN, q = (i / NN) % NN, b = i / (NN * NN);
  const float* lp = lhs2 + ((size_t)b * NN + m) * TT;
  const float* rp = rhs + ((size_t)b * NN + q) * TT;
  float s = 0.f;
  for (int t = 0; t < TT; ++t) s += lp[t] * rp[t];
  out[i] = f2bf(sigm(s + bs[m * NN + q]));
}

__global__ void k_cvtVs(const float* __restrict__ Vs, unsigned short* __restrict__ o) {
  int i = blockIdx.x * blockDim.x + threadIdx.x;
  if (i < NN * NN) o[i] = f2bf(Vs[i]);
}

// S1[b,n,q] = sum_m Vs[n,m]*sig[b,m,q]  (bf16 WMMA, f32 accumulate)
// A = Vs_bf row-major [n][m]; B = sigT [b][q][m] (transposed, m contiguous).
// One wave handles a 16x64 output strip: 1 A-frag reused across 4 B-frags.
__global__ void k_gemm_S(const unsigned short* __restrict__ A,   // [512][512]
                         const unsigned short* __restrict__ Bt,  // [B][512][512]
                         float* __restrict__ D) {                // [B][512][512]
  int lane = threadIdx.x;
  int tm = blockIdx.x, tn0 = blockIdx.y * 4, b = blockIdx.z;
  const unsigned short* Bp = Bt + (size_t)b * NN * NN;
  float* Dp = D + (size_t)b * NN * NN;
  int hi = lane >> 4;
  int arow = tm * 16 + (lane & 15);
  int lq = lane & 15;
  v8f acc0 = {}, acc1 = {}, acc2 = {}, acc3 = {};
  for (int kk = 0; kk < NN; kk += 32) {
    const unsigned short* ar = A + (size_t)arow * NN + kk + hi * 16;
    v16bf a = load_frag16(ar, ar + 8);
    const unsigned short* b0 = Bp + (size_t)((tn0 + 0) * 16 + lq) * NN + kk + hi * 16;
    const unsigned short* b1 = Bp + (size_t)((tn0 + 1) * 16 + lq) * NN + kk + hi * 16;
    const unsigned short* b2 = Bp + (size_t)((tn0 + 2) * 16 + lq) * NN + kk + hi * 16;
    const unsigned short* b3 = Bp + (size_t)((tn0 + 3) * 16 + lq) * NN + kk + hi * 16;
    v16bf f0 = load_frag16(b0, b0 + 8);
    v16bf f1 = load_frag16(b1, b1 + 8);
    v16bf f2 = load_frag16(b2, b2 + 8);
    v16bf f3 = load_frag16(b3, b3 + 8);
    acc0 = __builtin_amdgcn_wmma_f32_16x16x32_bf16(false, a, false, f0, (short)0, acc0, false, false);
    acc1 = __builtin_amdgcn_wmma_f32_16x16x32_bf16(false, a, false, f1, (short)0, acc1, false, false);
    acc2 = __builtin_amdgcn_wmma_f32_16x16x32_bf16(false, a, false, f2, (short)0, acc2, false, false);
    acc3 = __builtin_amdgcn_wmma_f32_16x16x32_bf16(false, a, false, f3, (short)0, acc3, false, false);
  }
  int rowbase = tm * 16 + hi * 8;
#pragma unroll
  for (int r = 0; r < 8; ++r) {
    size_t ro = (size_t)(rowbase + r) * NN;
    Dp[ro + (tn0 + 0) * 16 + lq] = acc0[r];
    Dp[ro + (tn0 + 1) * 16 + lq] = acc1[r];
    Dp[ro + (tn0 + 2) * 16 + lq] = acc2[r];
    Dp[ro + (tn0 + 3) * 16 + lq] = acc3[r];
  }
}

// in-place softmax over n for each (b,q) column of S[b,n,q]
__global__ void k_softmax_S(float* __restrict__ S) {
  int bq = blockIdx.x; int b = bq >> 9; int q = bq & (NN - 1);
  float* base = S + (size_t)b * NN * NN + q;
  int tid = threadIdx.x;
  __shared__ float red[128];
  float vals[4]; float mx = -1e30f;
  for (int i = 0; i < 4; ++i) {
    vals[i] = base[(size_t)(tid + i * 128) * NN];
    mx = fmaxf(mx, vals[i]);
  }
  red[tid] = mx; __syncthreads();
  for (int off = 64; off > 0; off >>= 1) {
    if (tid < off) red[tid] = fmaxf(red[tid], red[tid + off]);
    __syncthreads();
  }
  mx = red[0]; __syncthreads();
  float sm = 0.f;
  for (int i = 0; i < 4; ++i) { vals[i] = expf(vals[i] - mx); sm += vals[i]; }
  red[tid] = sm; __syncthreads();
  for (int off = 64; off > 0; off >>= 1) {
    if (tid < off) red[tid] += red[tid + off];
    __syncthreads();
  }
  float inv = 1.f / red[0];
  for (int i = 0; i < 4; ++i) base[(size_t)(tid + i * 128) * NN] = vals[i] * inv;
}

// ---------------- cheb graph conv ----------------
// A_bf[b][n][k*512+m] = bf16( cheb[k,m,n] * S[b,m,n] )
__global__ void k_build_A(const float* __restrict__ cheb, const float* __restrict__ S,
                          unsigned short* __restrict__ A) {
  int i = blockIdx.x * blockDim.x + threadIdx.x;
  if (i >= BB * NN * KM) return;
  int km = i % KM, n = (i / KM) % NN, b = i / (KM * NN);
  int k = km / NN, m = km % NN;
  float v = cheb[((size_t)k * NN + m) * NN + n] * S[((size_t)b * NN + m) * NN + n];
  A[i] = f2bf(v);
}

// Yt[b][f*24+t][k*512+m] = bf16( sum_c x[b,m,c,t]*Theta[k,c,f] )
// Stored TRANSPOSED (contraction index km contiguous).
__global__ void k_build_Y(const float* __restrict__ x, const float* __restrict__ Theta,
                          unsigned short* __restrict__ Y) {
  int i = blockIdx.x * blockDim.x + threadIdx.x;
  if (i >= BB * KM * FT2) return;
  int km = i % KM, ft = (i / KM) % FT2, b = i / (KM * FT2);
  int k = km / NN, m = km % NN, f = ft / TT, t = ft % TT;
  const float* xp = x + ((size_t)b * NN + m) * CC * TT + t;
  const float* th = Theta + (size_t)k * CC * FCC + f;
  float s = 0.f;
  for (int c = 0; c < CC; ++c) s += xp[(size_t)c * TT] * th[(size_t)c * FCC];
  Y[i] = f2bf(s);
}

// G[b,f,n,t] = relu( sum_km A_bf[b,n,km] * Yt[b,ft,km] )  (bf16 WMMA)
// One wave: 16x64 output strip, 1 A-frag x 4 B-frags per K step.
__global__ void k_gemm_cheb(const unsigned short* __restrict__ A,  // [B][512][1536]
                            const unsigned short* __restrict__ Yt, // [B][1536][1536]
                            float* __restrict__ G) {               // [B][64][512][24]
  int lane = threadIdx.x;
  int tm = blockIdx.x, tn0 = blockIdx.y * 4, b = blockIdx.z;
  const unsigned short* Ap = A + (size_t)b * NN * KM;
  const unsigned short* Yp = Yt + (size_t)b * KM * FT2;
  int hi = lane >> 4;
  int arow = tm * 16 + (lane & 15);
  int lq = lane & 15;
  v8f acc0 = {}, acc1 = {}, acc2 = {}, acc3 = {};
  for (int kk = 0; kk < KM; kk += 32) {
    const unsigned short* ar = Ap + (size_t)arow * KM + kk + hi * 16;
    v16bf a = load_frag16(ar, ar + 8);
    const unsigned short* b0 = Yp + (size_t)((tn0 + 0) * 16 + lq) * KM + kk + hi * 16;
    const unsigned short* b1 = Yp + (size_t)((tn0 + 1) * 16 + lq) * KM + kk + hi * 16;
    const unsigned short* b2 = Yp + (size_t)((tn0 + 2) * 16 + lq) * KM + kk + hi * 16;
    const unsigned short* b3 = Yp + (size_t)((tn0 + 3) * 16 + lq) * KM + kk + hi * 16;
    v16bf f0 = load_frag16(b0, b0 + 8);
    v16bf f1 = load_frag16(b1, b1 + 8);
    v16bf f2 = load_frag16(b2, b2 + 8);
    v16bf f3 = load_frag16(b3, b3 + 8);
    acc0 = __builtin_amdgcn_wmma_f32_16x16x32_bf16(false, a, false, f0, (short)0, acc0, false, false);
    acc1 = __builtin_amdgcn_wmma_f32_16x16x32_bf16(false, a, false, f1, (short)0, acc1, false, false);
    acc2 = __builtin_amdgcn_wmma_f32_16x16x32_bf16(false, a, false, f2, (short)0, acc2, false, false);
    acc3 = __builtin_amdgcn_wmma_f32_16x16x32_bf16(false, a, false, f3, (short)0, acc3, false, false);
  }
  int rowbase = tm * 16 + hi * 8;
#pragma unroll
  for (int u = 0; u < 4; ++u) {
    int col = (tn0 + u) * 16 + lq;
    int f = col / TT, t = col % TT;
    v8f acc = (u == 0) ? acc0 : (u == 1) ? acc1 : (u == 2) ? acc2 : acc3;
#pragma unroll
    for (int r = 0; r < 8; ++r) {
      int n = rowbase + r;
      G[(((size_t)b * FCC + f) * NN + n) * TT + t] = fmaxf(acc[r], 0.f);
    }
  }
}

// ---------------- time conv + residual + relu + layernorm ----------------
// block = 64 threads (one per output channel f), one block per (b,n,t)
__global__ void k_final(const float* __restrict__ G, const float* __restrict__ x,
                        const float* __restrict__ tcw, const float* __restrict__ tcb,
                        const float* __restrict__ rcw, const float* __restrict__ rcb,
                        const float* __restrict__ lng, const float* __restrict__ lnb,
                        float* __restrict__ out) {
  int id = blockIdx.x;
  int t = id % TT, n = (id / TT) % NN, b = id / (TT * NN);
  int f = threadIdx.x;
  // (1,3) time conv, pad (0,1)
  float acc = tcb[f];
  for (int c = 0; c < CC; ++c) {
    const float* gp = G + (((size_t)b * FCC + c) * NN + n) * TT + t;
    float x0 = (t > 0) ? gp[-1] : 0.f;
    float x1 = gp[0];
    float x2 = (t < TT - 1) ? gp[1] : 0.f;
    const float* w = tcw + ((size_t)f * FCC + c) * 3;
    acc += x0 * w[0] + x1 * w[1] + x2 * w[2];
  }
  // residual 1x1 conv on original x
  float r = rcb[f];
  const float* xp = x + ((size_t)b * NN + n) * CC * TT + t;
  for (int c = 0; c < CC; ++c) r += xp[(size_t)c * TT] * rcw[(size_t)f * CC + c];
  float z = fmaxf(acc + r, 0.f);
  // layernorm over channels
  __shared__ float s1[64], s2[64];
  s1[f] = z; s2[f] = z * z; __syncthreads();
  for (int off = 32; off > 0; off >>= 1) {
    if (f < off) { s1[f] += s1[f + off]; s2[f] += s2[f + off]; }
    __syncthreads();
  }
  float mu = s1[0] * (1.f / FTT);
  float var = s2[0] * (1.f / FTT) - mu * mu;
  float yv = (z - mu) * rsqrtf(var + LN_EPS) * lng[f] + lnb[f];
  out[(((size_t)b * NN + n) * FTT + f) * TT + t] = yv;  // [b,N,FT,T]
}

// ---------------- host launch ----------------
static inline int cdiv(int a, int b) { return (a + b - 1) / b; }

extern "C" void kernel_launch(void* const* d_in, const int* in_sizes, int n_in,
                              void* d_out, int out_size, void* d_ws, size_t ws_size,
                              hipStream_t stream) {
  (void)in_sizes; (void)n_in; (void)out_size; (void)ws_size;
  const float* x     = (const float*)d_in[0];
  const float* cheb  = (const float*)d_in[1];
  /* nodes (d_in[2]) is an identity permutation -> ignored */
  const float* U1    = (const float*)d_in[3];
  const float* U2    = (const float*)d_in[4];
  const float* U3    = (const float*)d_in[5];
  const float* be    = (const float*)d_in[6];
  const float* Ve    = (const float*)d_in[7];
  const float* W1    = (const float*)d_in[8];
  const float* W2    = (const float*)d_in[9];
  const float* W3    = (const float*)d_in[10];
  const float* bs    = (const float*)d_in[11];
  const float* Vs    = (const float*)d_in[12];
  const float* Theta = (const float*)d_in[13];
  const float* tcw   = (const float*)d_in[14];
  const float* tcb   = (const float*)d_in[15];
  const float* rcw   = (const float*)d_in[16];
  const float* rcb   = (const float*)d_in[17];
  const float* lng   = (const float*)d_in[18];
  const float* lnb   = (const float*)d_in[19];
  float* out = (float*)d_out;

  // carve workspace
  char* w = (char*)d_ws;
  size_t off = 0;
  auto carve = [&](size_t bytes) -> void* {
    void* p = w + off;
    off += (bytes + 255) & ~(size_t)255;
    return p;
  };
  float* xTAt   = (float*)carve((size_t)BB * NN * CC * TT * 4);
  float* lhs_t  = (float*)carve((size_t)BB * TT * CC * 4);
  float* lhs_t2 = (float*)carve((size_t)BB * TT * NN * 4);
  float* rhs_t  = (float*)carve((size_t)BB * NN * TT * 4);
  float* ps_t   = (float*)carve((size_t)BB * TT * TT * 4);
  float* Emat   = (float*)carve((size_t)BB * TT * TT * 4);
  float* lhs_s  = (float*)carve((size_t)BB * NN * CC * 4);
  float* lhs_s2 = (float*)carve((size_t)BB * NN * TT * 4);
  float* rhs_s  = (float*)carve((size_t)BB * NN * TT * 4);
  unsigned short* sigT = (unsigned short*)carve((size_t)BB * NN * NN * 2);
  unsigned short* VsBf = (unsigned short*)carve((size_t)NN * NN * 2);
  float* S      = (float*)carve((size_t)BB * NN * NN * 4);
  unsigned short* Abf = (unsigned short*)carve((size_t)BB * NN * KM * 2);
  unsigned short* Ybf = (unsigned short*)carve((size_t)BB * KM * FT2 * 2);
  float* G      = (float*)carve((size_t)BB * FCC * NN * TT * 4);

  // temporal attention
  k_lhs_t   <<<cdiv(BB * TT * CC, 256), 256, 0, stream>>>(x, U1, lhs_t);
  k_lhs_t2  <<<cdiv(BB * TT * NN, 256), 256, 0, stream>>>(lhs_t, U2, lhs_t2);
  k_rhs_t   <<<cdiv(BB * NN * TT, 256), 256, 0, stream>>>(x, U3, rhs_t);
  k_prodsig_t<<<cdiv(BB * TT * TT, 256), 256, 0, stream>>>(lhs_t2, rhs_t, be, ps_t);
  k_E       <<<cdiv(BB * TT, 64), 64, 0, stream>>>(Ve, ps_t, Emat);
  k_xTAt    <<<cdiv(BB * NN * CC * TT, 256), 256, 0, stream>>>(x, Emat, xTAt);

  // spatial attention
  k_lhs_s   <<<cdiv(BB * NN * CC, 256), 256, 0, stream>>>(xTAt, W1, lhs_s);
  k_lhs_s2  <<<cdiv(BB * NN * TT, 256), 256, 0, stream>>>(lhs_s, W2, lhs_s2);
  k_rhs_s   <<<cdiv(BB * NN * TT, 256), 256, 0, stream>>>(xTAt, W3, rhs_s);
  k_prodsig_s<<<cdiv(BB * NN * NN, 256), 256, 0, stream>>>(lhs_s2, rhs_s, bs, sigT);
  k_cvtVs   <<<cdiv(NN * NN, 256), 256, 0, stream>>>(Vs, VsBf);
  {
    dim3 g(NN / 16, NN / 16 / 4, BB);
    k_gemm_S<<<g, 32, 0, stream>>>(VsBf, sigT, S);
  }
  k_softmax_S<<<BB * NN, 128, 0, stream>>>(S);

  // cheb graph conv (big bf16 WMMA GEMM, K = 3*512 fused)
  k_build_A <<<cdiv(BB * NN * KM, 256), 256, 0, stream>>>(cheb, S, Abf);
  k_build_Y <<<cdiv(BB * KM * FT2, 256), 256, 0, stream>>>(x, Theta, Ybf);
  {
    dim3 g(NN / 16, FT2 / 16 / 4, BB);
    k_gemm_cheb<<<g, 32, 0, stream>>>(Abf, Ybf, G);
  }

  // time conv + residual + relu + layernorm
  k_final<<<BB * NN * TT, 64, 0, stream>>>(G, x, tcw, tcb, rcw, rcb, lng, lnb, out);
}
